// CifPredictorV3_47742856463059
// MI455X (gfx1250) — matrix operand order repaired and checked by reading, not compiled, not run on hardware
//
#include <hip/hip_runtime.h>
#include <hip/hip_bf16.h>
#include <math.h>

typedef unsigned short u16;
typedef unsigned int   u32;
typedef __attribute__((ext_vector_type(16))) __bf16 v16bf;
typedef __attribute__((ext_vector_type(8)))  float  v8f;
typedef __attribute__((ext_vector_type(8)))  u16    v8us;

#define B_   16
#define S_   512
#define D_   512
#define T3_  1536
#define H4_  2048
#define TZ_  513

#define SM1  1.0f
#define NT1  0.0f
#define TAIL 0.45f
#define SM2  0.25f
#define NT2  0.01f

// d_out layout (f32): embeds | token_num | alphas | us_alphas | us_peaks
#define EMB_OFF  0
#define TOK_OFF  (B_*TZ_*D_)            // 4202496
#define ALPH_OFF (TOK_OFF + B_)         // +16
#define USA_OFF  (ALPH_OFF + B_*TZ_)    // +8208
#define USP_OFF  (USA_OFF + B_*T3_)     // +24576

__device__ __forceinline__ u16 f2bf(float f){
  u32 u = __float_as_uint(f);
  u32 r = (u + 0x7FFFu + ((u >> 16) & 1u)) >> 16;
  return (u16)r;
}
__device__ __forceinline__ float bf2f(u16 h){ return __uint_as_float(((u32)h) << 16); }
__device__ __forceinline__ float sigm(float x){ return 1.0f / (1.0f + __expf(-x)); }

// ---- WMMA fragment loaders (bf16 16x16x32, wave32 ISA layouts) -------------
// A (16xK): lane L holds row M=L%16; K runs [kb*32+hi*8,+8) and [kb*32+16+hi*8,+8)
__device__ __forceinline__ size_t a_lane_off(int ld){
  int lane = threadIdx.x & 31;
  return (size_t)(lane & 15) * ld + (size_t)(lane >> 4) * 8;
}
// B (Kx16) stored transposed as [n][k]: lane L holds col N=L%16; K runs [kb*32+hi*8,+8) and [kb*32+16+hi*8,+8)
__device__ __forceinline__ size_t b_lane_off(int ld){
  int lane = threadIdx.x & 31;
  return (size_t)(lane & 15) * ld + (size_t)(lane >> 4) * 8;
}
__device__ __forceinline__ v16bf load_a_at(const u16* p){
  union { v8us h[2]; v16bf v; } u;
  u.h[0] = *(const v8us*)p;
  u.h[1] = *(const v8us*)(p + 16);
  return u.v;
}
__device__ __forceinline__ v16bf load_b_at(const u16* p){
  union { v8us h[2]; v16bf v; } u;
  u.h[0] = *(const v8us*)p;
  u.h[1] = *(const v8us*)(p + 16);
  return u.v;
}
#define WMMA_BF16(a,b,c) __builtin_amdgcn_wmma_f32_16x16x32_bf16(false,(a),false,(b),(short)0,(c),false,false)

// ---- packing / casting ------------------------------------------------------
__global__ void k_cast_bf16(const float* __restrict__ s, u16* __restrict__ d, int n){
  int i = blockIdx.x * blockDim.x + threadIdx.x;
  if (i < n) d[i] = f2bf(s[i]);
}
// hpad[b][r][i]: r in [0,514), rows 0 and 513 zero, r=1..512 -> hidden[b][r-1][i]
__global__ void k_hpad(const float* __restrict__ hid, u16* __restrict__ hp){
  int i = blockIdx.x * blockDim.x + threadIdx.x;
  if (i >= B_ * 514 * D_) return;
  int d = i & 511; int r = (i / 512) % 514; int b = i / (512 * 514);
  float v = (r >= 1 && r <= 512) ? hid[((size_t)b * S_ + (r - 1)) * D_ + d] : 0.0f;
  hp[i] = f2bf(v);
}
// conv_w [o][i][k] -> cw [o][k*512+i]
__global__ void k_pack_convw(const float* __restrict__ w, u16* __restrict__ cw){
  int i = blockIdx.x * blockDim.x + threadIdx.x;
  if (i >= D_ * 3 * D_) return;
  int o = i / 1536; int rest = i % 1536; int k = rest / 512; int din = rest % 512;
  cw[i] = f2bf(w[((size_t)o * D_ + din) * 3 + k]);
}
// up_w [i][o][k] -> uw [k][o][i]
__global__ void k_pack_upw(const float* __restrict__ w, u16* __restrict__ uw){
  int i = blockIdx.x * blockDim.x + threadIdx.x;
  if (i >= 3 * D_ * D_) return;
  int k = i / (512 * 512); int r = i % (512 * 512); int o = r / 512; int din = r % 512;
  uw[i] = f2bf(w[((size_t)din * D_ + o) * 3 + k]);
}

// ---- GEMM: conv1d(k=3,pad=1) + ReLU -> q [B*S][D] f32 ----------------------
// Software-pipelined: K-block kb+1's fragments load while kb's WMMAs execute.
__global__ __launch_bounds__(256) void k_gemm_conv(const u16* __restrict__ hpad,
    const u16* __restrict__ cw, const float* __restrict__ cb, float* __restrict__ q){
  int wid = blockIdx.x * 8 + (threadIdx.x >> 5);      // 4096 waves
  int mt = wid >> 3, nt = wid & 7;
  int m0 = mt * 16, n0 = nt * 64;
  int b = m0 >> 9, s0 = m0 & 511;
  const u16* ab = hpad + (size_t)(b * 514 + s0) * D_ + a_lane_off(D_);
  const u16* bb = cw + (size_t)n0 * 1536 + b_lane_off(1536);
  v8f acc0 = {}, acc1 = {}, acc2 = {}, acc3 = {};
  v16bf a  = load_a_at(ab);
  v16bf c0 = load_b_at(bb);
  v16bf c1 = load_b_at(bb + 16 * 1536);
  v16bf c2 = load_b_at(bb + 32 * 1536);
  v16bf c3 = load_b_at(bb + 48 * 1536);
  for (int kb = 0; kb < 47; ++kb){
    int kn = kb + 1;
    int oa = (kn >> 4) * D_ + (kn & 15) * 32;
    int ob = kn * 32;
    v16bf an = load_a_at(ab + oa);
    v16bf d0 = load_b_at(bb + ob);
    v16bf d1 = load_b_at(bb + 16 * 1536 + ob);
    v16bf d2 = load_b_at(bb + 32 * 1536 + ob);
    v16bf d3 = load_b_at(bb + 48 * 1536 + ob);
    acc0 = WMMA_BF16(a, c0, acc0);
    acc1 = WMMA_BF16(a, c1, acc1);
    acc2 = WMMA_BF16(a, c2, acc2);
    acc3 = WMMA_BF16(a, c3, acc3);
    a = an; c0 = d0; c1 = d1; c2 = d2; c3 = d3;
  }
  acc0 = WMMA_BF16(a, c0, acc0);
  acc1 = WMMA_BF16(a, c1, acc1);
  acc2 = WMMA_BF16(a, c2, acc2);
  acc3 = WMMA_BF16(a, c3, acc3);
  int lane = threadIdx.x & 31, col = lane & 15, hi = lane >> 4;
  v8f accs[4] = {acc0, acc1, acc2, acc3};
#pragma unroll
  for (int j = 0; j < 4; ++j){
    int n = n0 + j * 16 + col;
    float bias = cb[n];
#pragma unroll
    for (int v = 0; v < 8; ++v){
      int m = m0 + v + 8 * hi;
      float x = accs[j][v] + bias;
      q[(size_t)m * D_ + n] = x > 0.0f ? x : 0.0f;
    }
  }
}

// ---- GEMM: ConvTranspose1d(stride=3) -> xu [B][3S][D] bf16 (bias added) ----
__global__ __launch_bounds__(256) void k_gemm_up(const u16* __restrict__ hpad,
    const u16* __restrict__ uw, const float* __restrict__ ub, u16* __restrict__ xu){
  int wid = blockIdx.x * 8 + (threadIdx.x >> 5);      // 12288 waves
  int k = wid / 4096; int rem = wid % 4096;
  int mt = rem >> 3, nt = rem & 7;
  int m0 = mt * 16, n0 = nt * 64;
  int b = m0 >> 9, s0 = m0 & 511;
  const u16* ab = hpad + (size_t)(b * 514 + s0 + 1) * D_ + a_lane_off(D_);
  const u16* bb = uw + (size_t)(k * D_ + n0) * D_ + b_lane_off(D_);
  v8f acc0 = {}, acc1 = {}, acc2 = {}, acc3 = {};
  v16bf a  = load_a_at(ab);
  v16bf c0 = load_b_at(bb);
  v16bf c1 = load_b_at(bb + 16 * D_);
  v16bf c2 = load_b_at(bb + 32 * D_);
  v16bf c3 = load_b_at(bb + 48 * D_);
  for (int kb = 0; kb < 15; ++kb){
    int ob = (kb + 1) * 32;
    v16bf an = load_a_at(ab + ob);
    v16bf d0 = load_b_at(bb + ob);
    v16bf d1 = load_b_at(bb + 16 * D_ + ob);
    v16bf d2 = load_b_at(bb + 32 * D_ + ob);
    v16bf d3 = load_b_at(bb + 48 * D_ + ob);
    acc0 = WMMA_BF16(a, c0, acc0);
    acc1 = WMMA_BF16(a, c1, acc1);
    acc2 = WMMA_BF16(a, c2, acc2);
    acc3 = WMMA_BF16(a, c3, acc3);
    a = an; c0 = d0; c1 = d1; c2 = d2; c3 = d3;
  }
  acc0 = WMMA_BF16(a, c0, acc0);
  acc1 = WMMA_BF16(a, c1, acc1);
  acc2 = WMMA_BF16(a, c2, acc2);
  acc3 = WMMA_BF16(a, c3, acc3);
  int lane = threadIdx.x & 31, col = lane & 15, hi = lane >> 4;
  v8f accs[4] = {acc0, acc1, acc2, acc3};
#pragma unroll
  for (int j = 0; j < 4; ++j){
    int n = n0 + j * 16 + col;
    float bias = ub[n];
#pragma unroll
    for (int v = 0; v < 8; ++v){
      int s = s0 + v + 8 * hi;
      int t3 = 3 * s + k;
      xu[((size_t)b * T3_ + t3) * D_ + n] = f2bf(accs[j][v] + bias);
    }
  }
}

// ---- GEMM: input gates xg = xu @ w_ih^T + b, both dirs ---------------------
// xg layout: [t][n(2048)][b(16)] bf16 so the recurrent kernel reads each
// lane's 4 gate pre-activation groups as contiguous 16-byte vectors.
__global__ __launch_bounds__(256) void k_gemm_gates(const u16* __restrict__ xu,
    const u16* __restrict__ wihF, const u16* __restrict__ wihB,
    const float* __restrict__ bihF, const float* __restrict__ bhhF,
    const float* __restrict__ bihB, const float* __restrict__ bhhB,
    u16* __restrict__ xgF, u16* __restrict__ xgB){
  int wid = blockIdx.x * 8 + (threadIdx.x >> 5);      // 98304 waves
  int dir = wid / 49152; int rem = wid % 49152;
  int mt = rem >> 5, nt = rem & 31;
  int m0 = mt * 16, n0 = nt * 64;
  const u16* wih = dir ? wihB : wihF;
  const float* bih = dir ? bihB : bihF;
  const float* bhh = dir ? bhhB : bhhF;
  u16* xg = dir ? xgB : xgF;
  const u16* ab = xu + (size_t)m0 * D_ + a_lane_off(D_);
  const u16* bb = wih + (size_t)n0 * D_ + b_lane_off(D_);
  v8f acc0 = {}, acc1 = {}, acc2 = {}, acc3 = {};
  v16bf a  = load_a_at(ab);
  v16bf c0 = load_b_at(bb);
  v16bf c1 = load_b_at(bb + 16 * D_);
  v16bf c2 = load_b_at(bb + 32 * D_);
  v16bf c3 = load_b_at(bb + 48 * D_);
  for (int kb = 0; kb < 15; ++kb){
    int ob = (kb + 1) * 32;
    v16bf an = load_a_at(ab + ob);
    v16bf d0 = load_b_at(bb + ob);
    v16bf d1 = load_b_at(bb + 16 * D_ + ob);
    v16bf d2 = load_b_at(bb + 32 * D_ + ob);
    v16bf d3 = load_b_at(bb + 48 * D_ + ob);
    acc0 = WMMA_BF16(a, c0, acc0);
    acc1 = WMMA_BF16(a, c1, acc1);
    acc2 = WMMA_BF16(a, c2, acc2);
    acc3 = WMMA_BF16(a, c3, acc3);
    a = an; c0 = d0; c1 = d1; c2 = d2; c3 = d3;
  }
  acc0 = WMMA_BF16(a, c0, acc0);
  acc1 = WMMA_BF16(a, c1, acc1);
  acc2 = WMMA_BF16(a, c2, acc2);
  acc3 = WMMA_BF16(a, c3, acc3);
  int lane = threadIdx.x & 31, col = lane & 15, hi = lane >> 4;
  int bidx = m0 / T3_, t0 = m0 % T3_;
  v8f accs[4] = {acc0, acc1, acc2, acc3};
#pragma unroll
  for (int j = 0; j < 4; ++j){
    int n = n0 + j * 16 + col;
    float bias = bih[n] + bhh[n];
#pragma unroll
    for (int v = 0; v < 8; ++v){
      int t = t0 + v + 8 * hi;
      xg[((size_t)t * H4_ + n) * B_ + bidx] = f2bf(accs[j][v] + bias);
    }
  }
}

// ---- Persistent recurrent LSTM: one 1024-thread WG per direction -----------
// Wave w owns h-columns [w*16, w*16+16): computes i,f,g,o gate tiles for those
// columns so c stays in registers. h (bf16) is double-buffered in LDS (one
// barrier publishes it). W_hh B-fragments are rotated one K-block ahead so
// every WMMA waits only partially on loadcnt instead of draining to zero.
__global__ __launch_bounds__(1024) void k_lstm(const u16* __restrict__ xgF,
    const u16* __restrict__ xgB, const u16* __restrict__ whhF,
    const u16* __restrict__ whhB, const float* __restrict__ out2_w,
    float* __restrict__ uspart){
  int dir = blockIdx.x;
  const u16* xg  = dir ? xgB  : xgF;
  const u16* whh = dir ? whhB : whhF;
  const float* w2 = out2_w + dir * D_;

  __shared__ __align__(16) u16 hsh[2][B_ * D_];
  __shared__ float red[1024];

  int tid = threadIdx.x;
  int wv = tid >> 5, lane = tid & 31, col = lane & 15, hi = lane >> 4;
  int hcol = wv * 16 + col;

  size_t aoff = a_lane_off(D_);
  size_t boff = b_lane_off(D_);
  const u16* wb0 = whh + (size_t)(0 * D_ + wv * 16) * D_ + boff;
  const u16* wb1 = whh + (size_t)(1 * D_ + wv * 16) * D_ + boff;
  const u16* wb2 = whh + (size_t)(2 * D_ + wv * 16) * D_ + boff;
  const u16* wb3 = whh + (size_t)(3 * D_ + wv * 16) * D_ + boff;

  for (int i = tid; i < B_ * D_; i += 1024) hsh[0][i] = 0;
  v8f creg = {};
  __syncthreads();

  int p = 0;
  for (int step = 0; step < T3_; ++step){
    int t = dir ? (T3_ - 1 - step) : step;

    // issue this step's xg loads first: latency hides under the whole GEMM
    const u16* xr = xg + (size_t)t * (H4_ * B_);
    size_t xo = (size_t)8 * hi;
    v8us x0 = *(const v8us*)(xr + (((size_t)(0 * D_ + hcol)) << 4) + xo);
    v8us x1 = *(const v8us*)(xr + (((size_t)(1 * D_ + hcol)) << 4) + xo);
    v8us x2 = *(const v8us*)(xr + (((size_t)(2 * D_ + hcol)) << 4) + xo);
    v8us x3 = *(const v8us*)(xr + (((size_t)(3 * D_ + hcol)) << 4) + xo);

    const u16* ap = &hsh[p][0] + aoff;
    v8f acc0 = {}, acc1 = {}, acc2 = {}, acc3 = {};
    v16bf c0 = load_b_at(wb0);
    v16bf c1 = load_b_at(wb1);
    v16bf c2 = load_b_at(wb2);
    v16bf c3 = load_b_at(wb3);
#pragma unroll
    for (int kb = 0; kb < 15; ++kb){
      int ob = (kb + 1) * 32;
      v16bf d0 = load_b_at(wb0 + ob);
      v16bf d1 = load_b_at(wb1 + ob);
      v16bf d2 = load_b_at(wb2 + ob);
      v16bf d3 = load_b_at(wb3 + ob);
      v16bf a  = load_a_at(ap + kb * 32);
      acc0 = WMMA_BF16(a, c0, acc0);
      acc1 = WMMA_BF16(a, c1, acc1);
      acc2 = WMMA_BF16(a, c2, acc2);
      acc3 = WMMA_BF16(a, c3, acc3);
      c0 = d0; c1 = d1; c2 = d2; c3 = d3;
    }
    {
      v16bf a = load_a_at(ap + 15 * 32);
      acc0 = WMMA_BF16(a, c0, acc0);
      acc1 = WMMA_BF16(a, c1, acc1);
      acc2 = WMMA_BF16(a, c2, acc2);
      acc3 = WMMA_BF16(a, c3, acc3);
    }

    float hnew[8];
#pragma unroll
    for (int v = 0; v < 8; ++v){
      float gi = acc0[v] + bf2f(x0[v]);
      float gf = acc1[v] + bf2f(x1[v]);
      float gg = acc2[v] + bf2f(x2[v]);
      float go = acc3[v] + bf2f(x3[v]);
      float c  = sigm(gf) * creg[v] + sigm(gi) * tanhf(gg);
      creg[v] = c;
      hnew[v] = sigm(go) * tanhf(c);
    }

    u16* hn = &hsh[p ^ 1][0];        // write other buffer: no barrier needed
#pragma unroll
    for (int v = 0; v < 8; ++v)
      hn[(v + 8 * hi) * D_ + hcol] = f2bf(hnew[v]);
    __syncthreads();                 // publish new h (GEMM + head both read it)

    // fused out2 head: uspart[dir][t][b] = sum_k h[b][k] * w2[k]
    {
      int m = tid >> 6, seg = tid & 63;
      const u16* hr = hn + m * D_ + seg * 8;
      float pp = 0.0f;
#pragma unroll
      for (int k = 0; k < 8; ++k) pp += bf2f(hr[k]) * w2[seg * 8 + k];
      red[tid] = pp;
    }
    __syncthreads();
    if (tid < B_){
      float s = 0.0f;
      for (int i = 0; i < 64; ++i) s += red[tid * 64 + i];
      uspart[((size_t)dir * T3_ + t) * B_ + tid] = s;
    }
    p ^= 1;
  }
}

// ---- alphas head + tail ----------------------------------------------------
__global__ void k_alphas(const float* __restrict__ q, const float* __restrict__ ow,
    const float* __restrict__ ob, const float* __restrict__ mask, float* __restrict__ out){
  int idx = blockIdx.x * blockDim.x + threadIdx.x;
  if (idx >= B_ * TZ_) return;
  int b = idx / TZ_, s = idx % TZ_;
  const float* mk = mask + (size_t)b * S_;
  float a;
  if (s < S_){
    const float* qr = q + ((size_t)b * S_ + s) * D_;
    float dot = 0.0f;
    for (int i = 0; i < D_; ++i) dot += qr[i] * ow[i];
    a = sigm(dot + ob[0]);
    a = a * SM1 - NT1; a = a > 0.0f ? a : 0.0f;
    a *= mk[s];
    float tail = ((s == 0) ? 1.0f : mk[s - 1]) - mk[s];
    a += tail * TAIL;
  } else {
    a = mk[S_ - 1] * TAIL;
  }
  out[ALPH_OFF + (size_t)b * TZ_ + s] = a;
}

__global__ void k_token(float* __restrict__ out){
  __shared__ float r[256];
  int b = blockIdx.x, tid = threadIdx.x;
  float s = 0.0f;
  for (int t = tid; t < TZ_; t += 256) s += out[ALPH_OFF + (size_t)b * TZ_ + t];
  r[tid] = s; __syncthreads();
  for (int h = 128; h > 0; h >>= 1){ if (tid < h) r[tid] += r[tid + h]; __syncthreads(); }
  if (tid == 0) out[TOK_OFF + b] = floorf(r[0]);
}

__global__ void k_zero(float* __restrict__ p, size_t n){
  size_t i = (size_t)blockIdx.x * blockDim.x + threadIdx.x;
  size_t stride = (size_t)gridDim.x * blockDim.x;
  for (; i < n; i += stride) p[i] = 0.0f;
}

// ---- CIF sequential scan: one block per batch, thread d owns channel d -----
__global__ __launch_bounds__(512) void k_cif(const float* __restrict__ hid,
    float* __restrict__ out){
  int b = blockIdx.x, d = threadIdx.x;
  const float* al = out + ALPH_OFF + (size_t)b * TZ_;
  float integrate = 0.0f, frame = 0.0f;
  int n = 0;
  for (int t = 0; t < TZ_; ++t){
    float alpha = al[t];
    float ht = (t < S_) ? hid[((size_t)b * S_ + t) * D_ + d] : 0.0f;
    float dist = 1.0f - integrate;
    integrate += alpha;
    bool fire = integrate >= 1.0f;     // THRESH
    float cur = fire ? dist : alpha;
    float frame_out = frame + cur * ht;
    if (fire){
      if (n < TZ_) out[EMB_OFF + ((size_t)b * TZ_ + n) * D_ + d] = frame_out;
      n++;
      integrate -= 1.0f;
      frame = (alpha - cur) * ht;
    } else {
      frame = frame_out;
    }
  }
}

// ---- final us_alphas / us_peaks: one block per batch ------------------------
__global__ __launch_bounds__(256) void k_us_final(const float* __restrict__ uspart,
    const float* __restrict__ out2_b, const float* __restrict__ mask,
    float* __restrict__ out){
  __shared__ float usl[T3_];
  __shared__ float r[256];
  __shared__ float scale_sh;
  int b = blockIdx.x, tid = threadIdx.x;
  float loc = 0.0f;
  for (int t = tid; t < T3_; t += 256){
    float pf = uspart[(size_t)t * B_ + b];
    float pb = uspart[((size_t)T3_ + t) * B_ + b];
    float g = sigm(pf + pb + out2_b[0]);
    float u = g * SM2 - NT2; u = u > 0.0f ? u : 0.0f;
    u *= mask[(size_t)b * S_ + t / 3];
    usl[t] = u; loc += u;
  }
  r[tid] = loc; __syncthreads();
  for (int h = 128; h > 0; h >>= 1){ if (tid < h) r[tid] += r[tid + h]; __syncthreads(); }
  if (tid == 0){
    float tot = r[0];
    scale_sh = out[TOK_OFF + b] / tot;
  }
  __syncthreads();
  float sc = scale_sh;
  for (int t = tid; t < T3_; t += 256)
    out[USA_OFF + (size_t)b * T3_ + t] = usl[t] * sc;
  __syncthreads();
  if (tid == 0){
    float integ = 0.0f; const float th = 1.0f - 0.0001f;
    for (int t = 0; t < T3_; ++t){
      integ += usl[t] * sc;
      out[USP_OFF + (size_t)b * T3_ + t] = integ;
      if (integ >= th) integ -= th;
    }
  }
}

// ---- host-side orchestration ------------------------------------------------
extern "C" void kernel_launch(void* const* d_in, const int* in_sizes, int n_in,
                              void* d_out, int out_size, void* d_ws, size_t ws_size,
                              hipStream_t stream) {
  const float* hidden = (const float*)d_in[0];
  const float* mask   = (const float*)d_in[1];
  const float* conv_w = (const float*)d_in[2];
  const float* conv_b = (const float*)d_in[3];
  const float* out_w  = (const float*)d_in[4];
  const float* out_b  = (const float*)d_in[5];
  const float* up_w   = (const float*)d_in[6];
  const float* up_b   = (const float*)d_in[7];
  const float* w_ih_f = (const float*)d_in[8];
  const float* w_hh_f = (const float*)d_in[9];
  const float* b_ih_f = (const float*)d_in[10];
  const float* b_hh_f = (const float*)d_in[11];
  const float* w_ih_b = (const float*)d_in[12];
  const float* w_hh_b = (const float*)d_in[13];
  const float* b_ih_b = (const float*)d_in[14];
  const float* b_hh_b = (const float*)d_in[15];
  const float* out2_w = (const float*)d_in[16];
  const float* out2_b = (const float*)d_in[17];
  float* out = (float*)d_out;

  char* ws = (char*)d_ws;
  size_t off = 0;
  auto alloc = [&](size_t bytes) -> void* {
    void* p = ws + off;
    off += (bytes + 255) & ~(size_t)255;
    return p;
  };
  u16*   hpad = (u16*)  alloc((size_t)B_ * 514 * D_ * 2);
  u16*   cw   = (u16*)  alloc((size_t)D_ * 1536 * 2);
  u16*   uw   = (u16*)  alloc((size_t)3 * D_ * D_ * 2);
  u16*   wihF = (u16*)  alloc((size_t)H4_ * D_ * 2);
  u16*   whhF = (u16*)  alloc((size_t)H4_ * D_ * 2);
  u16*   wihB = (u16*)  alloc((size_t)H4_ * D_ * 2);
  u16*   whhB = (u16*)  alloc((size_t)H4_ * D_ * 2);
  float* q    = (float*)alloc((size_t)B_ * S_ * D_ * 4);
  u16*   xu   = (u16*)  alloc((size_t)B_ * T3_ * D_ * 2);
  u16*   xgF  = (u16*)  alloc((size_t)T3_ * B_ * H4_ * 2);
  u16*   xgB  = (u16*)  alloc((size_t)T3_ * B_ * H4_ * 2);
  float* usp  = (float*)alloc((size_t)2 * T3_ * B_ * 4);
  (void)ws_size; (void)n_in; (void)in_sizes; (void)out_size;

  // pack / cast
  {
    int n = B_ * 514 * D_;
    k_hpad<<<(n + 255) / 256, 256, 0, stream>>>(hidden, hpad);
  }
  k_pack_convw<<<(D_ * 1536 + 255) / 256, 256, 0, stream>>>(conv_w, cw);
  k_pack_upw<<<(3 * D_ * D_ + 255) / 256, 256, 0, stream>>>(up_w, uw);
  {
    int n = H4_ * D_;
    int g = (n + 255) / 256;
    k_cast_bf16<<<g, 256, 0, stream>>>(w_ih_f, wihF, n);
    k_cast_bf16<<<g, 256, 0, stream>>>(w_hh_f, whhF, n);
    k_cast_bf16<<<g, 256, 0, stream>>>(w_ih_b, wihB, n);
    k_cast_bf16<<<g, 256, 0, stream>>>(w_hh_b, whhB, n);
  }

  // alphas branch
  k_gemm_conv<<<512, 256, 0, stream>>>(hpad, cw, conv_b, q);
  k_alphas<<<(B_ * TZ_ + 255) / 256, 256, 0, stream>>>(q, out_w, out_b, mask, out);
  k_token<<<B_, 256, 0, stream>>>(out);
  k_zero<<<4096, 256, 0, stream>>>(out + EMB_OFF, (size_t)B_ * TZ_ * D_);
  k_cif<<<B_, 512, 0, stream>>>(hidden, out);

  // timestamp branch
  k_gemm_up<<<1536, 256, 0, stream>>>(hpad, uw, up_b, xu);
  k_gemm_gates<<<12288, 256, 0, stream>>>(xu, wihF, wihB, b_ih_f, b_hh_f,
                                          b_ih_b, b_hh_b, xgF, xgB);
  k_lstm<<<2, 1024, 0, stream>>>(xgF, xgB, whhF, whhB, out2_w, usp);
  k_us_final<<<B_, 256, 0, stream>>>(usp, out2_b, mask, out);
}
